// MACE_65017214927004
// MI455X (gfx1250) — matrix operand order repaired and checked by reading, not compiled, not run on hardware
//
#include <hip/hip_runtime.h>

typedef __attribute__((ext_vector_type(16))) _Float16 v16h;
typedef __attribute__((ext_vector_type(8)))  float    v8f;

#define GF_TRANSB 1
#define GF_ACCUM  2

__device__ __forceinline__ float siluf(float x){
  float s = 1.0f/(1.0f+__expf(-x)); return x*s;
}
__device__ __forceinline__ float dsiluf(float x){
  float s = 1.0f/(1.0f+__expf(-x)); return s*(1.0f+x*(1.0f-s));
}

// ============================================================================
// Templated WMMA GEMM:  C[M,Nt] (+)= A[M,Kt] * B     (f16 in, f32 accum)
// B is [Kt,Nt] row-major, or [Nt,Kt] row-major when GF_TRANSB (B^T used).
// Requirements (guaranteed by caller): M % 128 == 0, Nt % 16 == 0, Kt % 32 == 0.
// wave32; 8 waves/block, one 16-row tile per wave. B staged in LDS as [N][K] f16.
// Fragment k order: contiguous, k = 16*hsel + s, in both A and B.
// ============================================================================
template<int Nt, int Kt>
__global__ void k_wmma_gemm(const float* __restrict__ A,
                            const float* __restrict__ B,
                            float* __restrict__ C,
                            int M, int flags)
{
  __shared__ _Float16 ldsBT[Nt*Kt];
  for (int idx = threadIdx.x; idx < Nt*Kt; idx += 256) {
    int n = idx / Kt, k = idx - n*Kt;
    float v = (flags & GF_TRANSB) ? B[(size_t)n*Kt + k] : B[(size_t)k*Nt + n];
    ldsBT[idx] = (_Float16)v;
  }
  __syncthreads();

  const int wave = threadIdx.x >> 5;
  const int lane = threadIdx.x & 31;
  const int row0 = (blockIdx.x * 8 + wave) * 16;
  if (row0 >= M) return;
  const int hsel = lane >> 4;               // 0: lanes 0-15, 1: lanes 16-31
  const int lr   = lane & 15;

  // A fragments for this 16-row tile: row = row0+lr.
  // Per K-tile of 32, each lane holds one contiguous 16-float chunk:
  //   [k0 + hsel*16 .. +15]
  const float* ap = A + (size_t)(row0 + lr) * Kt;
  __builtin_prefetch(ap + (size_t)128*Kt, 0, 1);   // next row-block (speculative)
  v16h afr[Kt/32];
#pragma unroll
  for (int kk = 0; kk < Kt/32; ++kk) {
    const float* p0 = ap + kk*32 + hsel*16;
    float4 c0 = *(const float4*)(p0);
    float4 c1 = *(const float4*)(p0 + 4);
    float4 c2 = *(const float4*)(p0 + 8);
    float4 c3 = *(const float4*)(p0 + 12);
    v16h af;
    af[0]=(_Float16)c0.x;  af[1]=(_Float16)c0.y;  af[2]=(_Float16)c0.z;  af[3]=(_Float16)c0.w;
    af[4]=(_Float16)c1.x;  af[5]=(_Float16)c1.y;  af[6]=(_Float16)c1.z;  af[7]=(_Float16)c1.w;
    af[8]=(_Float16)c2.x;  af[9]=(_Float16)c2.y;  af[10]=(_Float16)c2.z; af[11]=(_Float16)c2.w;
    af[12]=(_Float16)c3.x; af[13]=(_Float16)c3.y; af[14]=(_Float16)c3.z; af[15]=(_Float16)c3.w;
    afr[kk] = af;
  }

  for (int n0 = 0; n0 < Nt; n0 += 16) {
    v8f acc = {0.f,0.f,0.f,0.f,0.f,0.f,0.f,0.f};
    if (flags & GF_ACCUM) {
#pragma unroll
      for (int r = 0; r < 8; ++r)
        acc[r] = C[(size_t)(row0 + r + hsel*8)*Nt + n0 + lr];
    }
#pragma unroll
    for (int kk = 0; kk < Kt/32; ++kk) {
      // B fragment: lane -> column n, 16 contiguous K halfs (hsel*16 offset)
      const _Float16* bp = ldsBT + (size_t)(n0 + lr)*Kt + kk*32 + hsel*16;
      v16h bf = *(const v16h*)bp;
      acc = __builtin_amdgcn_wmma_f32_16x16x32_f16(false, afr[kk], false, bf,
                                                   (short)0, acc, false, false);
    }
#pragma unroll
    for (int r = 0; r < 8; ++r)
      C[(size_t)(row0 + r + hsel*8)*Nt + n0 + lr] = acc[r];
  }
}

// Zero-pad Wr1 [8,64] -> [32,64]
__global__ void k_pad_w1(const float* __restrict__ Wr1, float* __restrict__ Wp)
{
  int t = blockIdx.x*blockDim.x + threadIdx.x;
  if (t >= 32*64) return;
  int k = t >> 6;
  Wp[t] = (k < 8) ? Wr1[t] : 0.f;
}

// ============================================================================
// Edge geometry + radial basis (forward).  edge_feats stored [E,32] zero-padded.
// ============================================================================
__global__ void k_edge_geom(const float* __restrict__ pos,
                            const float* __restrict__ shifts,
                            const int* __restrict__ ei,
                            float* __restrict__ y1, float* __restrict__ len,
                            float* __restrict__ ef, int E)
{
  int e = blockIdx.x*blockDim.x + threadIdx.x;
  if (e >= E) return;
  int s = ei[e], r = ei[E+e];
  float vx = pos[r*3+0]-pos[s*3+0]+shifts[e*3+0];
  float vy = pos[r*3+1]-pos[s*3+1]+shifts[e*3+1];
  float vz = pos[r*3+2]-pos[s*3+2]+shifts[e*3+2];
  float l  = sqrtf(vx*vx+vy*vy+vz*vz + 1e-12f);
  float inv = 1.0f/l;
  const float SQ3 = 1.7320508075688772f;
  y1[e*3+0]=SQ3*vx*inv; y1[e*3+1]=SQ3*vy*inv; y1[e*3+2]=SQ3*vz*inv;
  len[e]=l;
  float x = l*0.2f;                         // l / R_MAX
  float cut = 0.f;
  if (x < 1.f) {
    float x2=x*x, x4=x2*x2, x5=x4*x, x6=x5*x, x7=x6*x;
    cut = 1.f - 21.f*x5 + 35.f*x6 - 15.f*x7;
  }
  const float pref = 0.6324555320336759f;   // sqrt(2/R)
  const float PIf = 3.14159265358979f;
  float* efe = ef + (size_t)e*32;
#pragma unroll
  for (int b = 1; b <= 8; ++b) {
    float sb = __sinf(PIf*(float)b*x);
    efe[b-1] = pref*sb*inv*cut;
  }
#pragma unroll
  for (int j = 8; j < 32; ++j) efe[j] = 0.f;
}

// Edge-geometry backward: d(edge_feats)/dpos + d(Y1)/dpos -> forces
__global__ void k_edge_geom_bwd(const int* __restrict__ ei,
                                const float* __restrict__ y1,
                                const float* __restrict__ len,
                                const float* __restrict__ def,
                                const float* __restrict__ dy1,
                                float* __restrict__ forces, int E)
{
  int e = blockIdx.x*blockDim.x + threadIdx.x;
  if (e >= E) return;
  const float SQ3 = 1.7320508075688772f, ISQ3 = 0.5773502691896258f;
  float l = len[e], inv = 1.0f/l;
  float ux=y1[e*3+0]*ISQ3, uy=y1[e*3+1]*ISQ3, uz=y1[e*3+2]*ISQ3;
  float x = l*0.2f;
  float cut = 0.f, dcutdl = 0.f;
  if (x < 1.f) {
    float x2=x*x, x4=x2*x2, x5=x4*x, x6=x5*x, x7=x6*x;
    cut = 1.f - 21.f*x5 + 35.f*x6 - 15.f*x7;
    dcutdl = (-105.f*x4 + 210.f*x5 - 105.f*x6)*0.2f;
  }
  const float pref = 0.6324555320336759f;
  const float PIf = 3.14159265358979f;
  const float* defe = def + (size_t)e*32;
  float dlen = 0.f;
#pragma unroll
  for (int b = 1; b <= 8; ++b) {
    float arg = PIf*(float)b*x;
    float sb = __sinf(arg), cb = __cosf(arg);
    float dfb = pref*((PIf*(float)b*0.2f)*cb*inv - sb*inv*inv)*cut
              + pref*sb*inv*dcutdl;
    dlen += defe[b-1] * dfb;
  }
  float dux=SQ3*dy1[e*3+0], duy=SQ3*dy1[e*3+1], duz=SQ3*dy1[e*3+2];
  float dot = ux*dux+uy*duy+uz*duz;
  float gx = (dux-ux*dot)*inv + ux*dlen;    // d total / d vec
  float gy = (duy-uy*dot)*inv + uy*dlen;
  float gz = (duz-uz*dot)*inv + uz*dlen;
  int s = ei[e], r = ei[E+e];
  atomicAdd(&forces[r*3+0], -gx); atomicAdd(&forces[r*3+1], -gy); atomicAdd(&forces[r*3+2], -gz);
  atomicAdd(&forces[s*3+0],  gx); atomicAdd(&forces[s*3+1],  gy); atomicAdd(&forces[s*3+2],  gz);
}

// ============================================================================
// Node embedding + fixed atomic energies
// ============================================================================
__global__ void k_embed(const int* __restrict__ sp, const float* __restrict__ Wemb,
                        float* __restrict__ s0, int N)
{
  int t = blockIdx.x*blockDim.x + threadIdx.x;
  if (t >= (N<<6)) return;
  int n = t>>6, c = t&63;
  s0[t] = Wemb[sp[n]*64 + c];
}

__global__ void k_e0(const int* __restrict__ sp, const int* __restrict__ batch,
                     const float* __restrict__ AE, float* __restrict__ contrib, int N)
{
  int n = blockIdx.x*blockDim.x + threadIdx.x;
  if (n >= N) return;
  atomicAdd(&contrib[batch[n]*3 + 0], AE[sp[n]]);
}

// ============================================================================
// SiLU fwd/bwd (elementwise)
// ============================================================================
__global__ void k_silu_fwd(const float* __restrict__ z, float* __restrict__ a, int n){
  int t = blockIdx.x*blockDim.x + threadIdx.x;
  if (t < n) a[t] = siluf(z[t]);
}
__global__ void k_silu_bwd(const float* __restrict__ z, const float* __restrict__ da,
                           float* __restrict__ dz, int n){
  int t = blockIdx.x*blockDim.x + threadIdx.x;
  if (t < n) dz[t] = da[t]*dsiluf(z[t]);
}

// ============================================================================
// Depthwise TP messages + scatter (forward);   1/AVG_NEI folded in
// ============================================================================
__global__ void k_message_fwd(const int* __restrict__ ei,
                              const float* __restrict__ su, const float* __restrict__ vu,
                              const float* __restrict__ y1, const float* __restrict__ w,
                              float* __restrict__ ags, float* __restrict__ agv,
                              int E, int NC)
{
  int t = blockIdx.x*blockDim.x + threadIdx.x;
  if (t >= (E<<6)) return;
  int e = t>>6, c = t&63;
  int s = ei[e], r = ei[E+e];
  const float* wp = w + (size_t)e*256 + c*4;
  float w0=wp[0], w1=wp[1], w2=wp[2], w3=wp[3];
  float se  = su[s*64+c];
  float vex = vu[s*64+c], vey = vu[NC+s*64+c], vez = vu[2*NC+s*64+c];
  float yx=y1[e*3+0], yy=y1[e*3+1], yz=y1[e*3+2];
  float ms = w0*se + w1*(vex*yx+vey*yy+vez*yz);
  float k2 = w2*se;
  const float inv = 0.03125f;               // 1/AVG_NEI
  atomicAdd(&ags[r*64+c],        ms*inv);
  atomicAdd(&agv[r*64+c],        (k2*yx + w3*vex)*inv);
  atomicAdd(&agv[NC+r*64+c],     (k2*yy + w3*vey)*inv);
  atomicAdd(&agv[2*NC+r*64+c],   (k2*yz + w3*vez)*inv);
}

// Message backward: per-edge grads for w, Y1, and sender su/vu
__global__ void k_message_bwd(const int* __restrict__ ei,
                              const float* __restrict__ gags, const float* __restrict__ gagv,
                              const float* __restrict__ su, const float* __restrict__ vu,
                              const float* __restrict__ y1, const float* __restrict__ w,
                              float* __restrict__ gsu, float* __restrict__ gvu,
                              float* __restrict__ dw, float* __restrict__ dy1,
                              int E, int NC)
{
  int t = blockIdx.x*blockDim.x + threadIdx.x;
  if (t >= (E<<6)) return;
  int e = t>>6, c = t&63;
  int s = ei[e], r = ei[E+e];
  const float inv = 0.03125f;
  float gms = gags[r*64+c]*inv;
  float gmx = gagv[r*64+c]*inv, gmy = gagv[NC+r*64+c]*inv, gmz = gagv[2*NC+r*64+c]*inv;
  const float* wp = w + (size_t)e*256 + c*4;
  float w0=wp[0], w1=wp[1], w2=wp[2], w3=wp[3];
  float se  = su[s*64+c];
  float vex = vu[s*64+c], vey = vu[NC+s*64+c], vez = vu[2*NC+s*64+c];
  float yx=y1[e*3+0], yy=y1[e*3+1], yz=y1[e*3+2];
  float dvy   = vex*yx+vey*yy+vez*yz;
  float gdoty = gmx*yx+gmy*yy+gmz*yz;
  float* dwp = dw + (size_t)e*256 + c*4;
  dwp[0] = gms*se;
  dwp[1] = gms*dvy;
  dwp[2] = se*gdoty;
  dwp[3] = gmx*vex+gmy*vey+gmz*vez;
  atomicAdd(&gsu[s*64+c], gms*w0 + w2*gdoty);
  atomicAdd(&gvu[s*64+c],      gms*w1*yx + gmx*w3);
  atomicAdd(&gvu[NC+s*64+c],   gms*w1*yy + gmy*w3);
  atomicAdd(&gvu[2*NC+s*64+c], gms*w1*yz + gmz*w3);
  atomicAdd(&dy1[e*3+0], gms*w1*vex + gmx*w2*se);
  atomicAdd(&dy1[e*3+1], gms*w1*vey + gmy*w2*se);
  atomicAdd(&dy1[e*3+2], gms*w1*vez + gmz*w2*se);
}

// ============================================================================
// Equivariant product basis (correlation-3, l<=1) fwd/bwd
// ============================================================================
__global__ void k_product_fwd(const int* __restrict__ sp,
                              const float* __restrict__ fs, const float* __restrict__ fv,
                              const float* __restrict__ Wps, const float* __restrict__ Wpv,
                              float* __restrict__ tmp_s, float* __restrict__ vtmp,
                              int N, int NC, int last)
{
  int t = blockIdx.x*blockDim.x + threadIdx.x;
  if (t >= (N<<6)) return;
  int n = t>>6, c = t&63, z = sp[n];
  float f = fs[t];
  float vx=fv[t], vy=fv[NC+t], vz=fv[2*NC+t];
  float vn2 = vx*vx+vy*vy+vz*vz;
  const float* ws = Wps + z*320 + c;        // [Z,5,C], stride 64 between k
  tmp_s[t] = f*ws[0] + f*f*ws[64] + vn2*ws[128] + f*f*f*ws[192] + f*vn2*ws[256];
  if (!last) {
    const float* wv = Wpv + z*192 + c;      // [Z,3,C]
    float q = wv[0] + wv[64]*f + wv[128]*f*f;
    vtmp[t]=vx*q; vtmp[NC+t]=vy*q; vtmp[2*NC+t]=vz*q;
  }
}

__global__ void k_product_bwd(const int* __restrict__ sp,
                              const float* __restrict__ gtmp, const float* __restrict__ gvtmp,
                              const float* __restrict__ fs, const float* __restrict__ fv,
                              const float* __restrict__ Wps, const float* __restrict__ Wpv,
                              float* __restrict__ gfs, float* __restrict__ gfv,
                              int N, int NC, int last)
{
  int t = blockIdx.x*blockDim.x + threadIdx.x;
  if (t >= (N<<6)) return;
  int n = t>>6, c = t&63, z = sp[n];
  float gt = gtmp[t];
  float f = fs[t];
  float vx=fv[t], vy=fv[NC+t], vz=fv[2*NC+t];
  float vn2 = vx*vx+vy*vy+vz*vz;
  const float* ws = Wps + z*320 + c;
  float gf   = gt*(ws[0] + 2.f*f*ws[64] + 3.f*f*f*ws[192] + vn2*ws[256]);
  float gvn2 = gt*(ws[128] + f*ws[256]);
  float gx = 2.f*vx*gvn2, gy = 2.f*vy*gvn2, gz = 2.f*vz*gvn2;
  if (!last) {
    const float* wv = Wpv + z*192 + c;
    float q = wv[0] + wv[64]*f + wv[128]*f*f;
    float ax=gvtmp[t], ay=gvtmp[NC+t], az=gvtmp[2*NC+t];
    gx += ax*q; gy += ay*q; gz += az*q;
    gf += (ax*vx+ay*vy+az*vz)*(wv[64] + 2.f*wv[128]*f);
  }
  gfs[t]=gf; gfv[t]=gx; gfv[NC+t]=gy; gfv[2*NC+t]=gz;
}

// ============================================================================
// Element-dependent skip connections (einsum with one-hot == per-node gather)
// ============================================================================
__global__ void k_sc_add_s(const int* __restrict__ sp, const float* __restrict__ s_in,
                           const float* __restrict__ Wsc, float* __restrict__ s_out, int N)
{
  int t = blockIdx.x*blockDim.x + threadIdx.x;
  if (t >= (N<<6)) return;
  int n = t>>6, d = t&63;
  const float* W = Wsc + sp[n]*4096 + d;    // [Z,C,C], stride 64 over c
  const float* si = s_in + n*64;
  float acc = 0.f;
  for (int c = 0; c < 64; ++c) acc += si[c]*W[c*64];
  s_out[t] += acc;
}
__global__ void k_sc_bwd_s(const int* __restrict__ sp, const float* __restrict__ gs,
                           const float* __restrict__ Wsc, float* __restrict__ gs_in, int N)
{
  int t = blockIdx.x*blockDim.x + threadIdx.x;
  if (t >= (N<<6)) return;
  int n = t>>6, c = t&63;
  const float* W = Wsc + sp[n]*4096 + c*64; // contiguous over d
  const float* g = gs + n*64;
  float acc = 0.f;
  for (int d = 0; d < 64; ++d) acc += g[d]*W[d];
  gs_in[t] += acc;
}
__global__ void k_sc_add_v(const int* __restrict__ sp, const float* __restrict__ v_in,
                           const float* __restrict__ Wsc, float* __restrict__ v_out,
                           int N, int NC)
{
  int t = blockIdx.x*blockDim.x + threadIdx.x;
  if (t >= (N<<6)) return;
  int n = t>>6, d = t&63;
  const float* W = Wsc + sp[n]*4096 + d;
  float ax=0.f, ay=0.f, az=0.f;
  for (int c = 0; c < 64; ++c) {
    float wv = W[c*64];
    ax += v_in[n*64+c]*wv; ay += v_in[NC+n*64+c]*wv; az += v_in[2*NC+n*64+c]*wv;
  }
  v_out[t] += ax; v_out[NC+t] += ay; v_out[2*NC+t] += az;
}
__global__ void k_sc_bwd_v(const int* __restrict__ sp, const float* __restrict__ gv,
                           const float* __restrict__ Wsc, float* __restrict__ gv_in,
                           int N, int NC)
{
  int t = blockIdx.x*blockDim.x + threadIdx.x;
  if (t >= (N<<6)) return;
  int n = t>>6, c = t&63;
  const float* W = Wsc + sp[n]*4096 + c*64;
  float ax=0.f, ay=0.f, az=0.f;
  for (int d = 0; d < 64; ++d) {
    float wv = W[d];
    ax += gv[n*64+d]*wv; ay += gv[NC+n*64+d]*wv; az += gv[2*NC+n*64+d]*wv;
  }
  gv_in[t] += ax; gv_in[NC+t] += ay; gv_in[2*NC+t] += az;
}

// ============================================================================
// Readouts (fwd + bwd)
// ============================================================================
__global__ void k_readout_lin(const float* __restrict__ s, const float* __restrict__ Wro,
                              const int* __restrict__ batch, float* __restrict__ contrib,
                              int col, int N)
{
  int n = blockIdx.x*blockDim.x + threadIdx.x;
  if (n >= N) return;
  const float* sn = s + n*64;
  float e = 0.f;
  for (int c = 0; c < 64; ++c) e += sn[c]*Wro[c];
  atomicAdd(&contrib[batch[n]*3 + col], e);
}
__global__ void k_readout_lin_bwd(const float* __restrict__ Wro, float* __restrict__ gs, int N)
{
  int t = blockIdx.x*blockDim.x + threadIdx.x;
  if (t >= (N<<6)) return;
  gs[t] += Wro[t & 63];
}
__global__ void k_readout_nl(const float* __restrict__ s, const float* __restrict__ W1,
                             const float* __restrict__ W2, const int* __restrict__ batch,
                             float* __restrict__ contrib, int col, int N)
{
  int n = blockIdx.x*blockDim.x + threadIdx.x;
  if (n >= N) return;
  const float* sn = s + n*64;
  float e = 0.f;
  for (int j = 0; j < 16; ++j) {
    float h = 0.f;
    for (int c = 0; c < 64; ++c) h += sn[c]*W1[c*16+j];
    e += siluf(h)*W2[j];
  }
  atomicAdd(&contrib[batch[n]*3 + col], e);
}
__global__ void k_readout_nl_bwd(const float* __restrict__ s, const float* __restrict__ W1,
                                 const float* __restrict__ W2, float* __restrict__ gs, int N)
{
  int n = blockIdx.x*blockDim.x + threadIdx.x;
  if (n >= N) return;
  const float* sn = s + n*64;
  float tj[16];
  for (int j = 0; j < 16; ++j) {
    float h = 0.f;
    for (int c = 0; c < 64; ++c) h += sn[c]*W1[c*16+j];
    tj[j] = dsiluf(h)*W2[j];
  }
  for (int c = 0; c < 64; ++c) {
    float g = 0.f;
    for (int j = 0; j < 16; ++j) g += W1[c*16+j]*tj[j];
    gs[n*64+c] += g;
  }
}

__global__ void k_total(float* __restrict__ total, const float* __restrict__ contrib, int G)
{
  int g = threadIdx.x;
  if (g < G) total[g] = contrib[g*3+0] + contrib[g*3+1] + contrib[g*3+2];
}

// ============================================================================
// Host orchestration
// ============================================================================
extern "C" void kernel_launch(void* const* d_in, const int* in_sizes, int n_in,
                              void* d_out, int out_size, void* d_ws, size_t ws_size,
                              hipStream_t stream)
{
  (void)in_sizes; (void)n_in; (void)ws_size;
  const int N = 8192, E = 262144, G = 16;
  const int NC = N*64;
  const int T = 256;

  const float* pos    = (const float*)d_in[0];
  const int*   sp     = (const int*)  d_in[1];
  const int*   ei     = (const int*)  d_in[2];
  const int*   batch  = (const int*)  d_in[3];
  const float* shifts = (const float*)d_in[4];
  // Params flattened in JAX pytree (sorted-key) order.
  const float* Wemb   = (const float*)d_in[5];
  const float* AE     = (const float*)d_in[6];
  const float *Wo0[2], *Wo1[2], *Wps[2], *Wpv[2], *Wpl0[2], *Wpl1[2];
  const float *Wr1[2], *Wr2[2], *Wr3[2], *Wsc0[2], *Wsc1[2], *Wup0[2], *Wup1[2];
  const float *WroL = nullptr, *WroA = nullptr, *WroB = nullptr;
  int i = 7;
  // layer 0: Wo0 Wo1 Wp_s Wp_v Wpl0 Wpl1 Wr1 Wr2 Wr3 Wro Wsc0 Wsc1 Wup0 Wup1
  Wo0[0]=(const float*)d_in[i++]; Wo1[0]=(const float*)d_in[i++];
  Wps[0]=(const float*)d_in[i++]; Wpv[0]=(const float*)d_in[i++];
  Wpl0[0]=(const float*)d_in[i++]; Wpl1[0]=(const float*)d_in[i++];
  Wr1[0]=(const float*)d_in[i++]; Wr2[0]=(const float*)d_in[i++]; Wr3[0]=(const float*)d_in[i++];
  WroL=(const float*)d_in[i++];
  Wsc0[0]=(const float*)d_in[i++]; Wsc1[0]=(const float*)d_in[i++];
  Wup0[0]=(const float*)d_in[i++]; Wup1[0]=(const float*)d_in[i++];
  // layer 1: ... Wr3 Wro1 Wro2 Wsc0 ...
  Wo0[1]=(const float*)d_in[i++]; Wo1[1]=(const float*)d_in[i++];
  Wps[1]=(const float*)d_in[i++]; Wpv[1]=(const float*)d_in[i++];
  Wpl0[1]=(const float*)d_in[i++]; Wpl1[1]=(const float*)d_in[i++];
  Wr1[1]=(const float*)d_in[i++]; Wr2[1]=(const float*)d_in[i++]; Wr3[1]=(const float*)d_in[i++];
  WroA=(const float*)d_in[i++]; WroB=(const float*)d_in[i++];
  Wsc0[1]=(const float*)d_in[i++]; Wsc1[1]=(const float*)d_in[i++];
  Wup0[1]=(const float*)d_in[i++]; Wup1[1]=(const float*)d_in[i++];

  // Outputs: total[16] ++ contrib[16,3] ++ forces[8192,3]
  float* out     = (float*)d_out;
  float* total   = out;
  float* contrib = out + G;
  float* forces  = out + G + G*3;
  hipMemsetAsync(d_out, 0, (size_t)out_size*sizeof(float), stream);

  // Workspace bump allocator (all f32)
  float* wsf = (float*)d_ws;
  size_t off = 0;
  auto AL = [&](size_t nfl){ float* p = wsf + off; off += nfl; return p; };
  float* ef   = AL((size_t)E*32);           // [E,32] zero-padded edge feats
  float* y1   = AL((size_t)E*3);
  float* len  = AL((size_t)E);
  float* z1   = AL((size_t)E*64);
  float* a1   = AL((size_t)E*64);
  float* z2   = AL((size_t)E*64);
  float* a2   = AL((size_t)E*64);
  float* wb[2]  = { AL((size_t)E*256), AL((size_t)E*256) };
  float* dwb  = AL((size_t)E*256);
  float* defb = AL((size_t)E*32);           // [E,32] padded d(edge_feats)
  float* dy1b = AL((size_t)E*3);
  float* Wr1p[2] = { AL(32*64), AL(32*64) };
  float* s0 = AL(NC); float* s1 = AL(NC); float* s2 = AL(NC);
  float* v0 = AL((size_t)3*NC); float* v1 = AL((size_t)3*NC);
  float* su[2]  = { AL(NC), AL(NC) };
  float* vu[2]  = { AL((size_t)3*NC), AL((size_t)3*NC) };
  float* fsb[2] = { AL(NC), AL(NC) };
  float* fvb[2] = { AL((size_t)3*NC), AL((size_t)3*NC) };
  float* ags = AL(NC); float* agv = AL((size_t)3*NC);
  float* tmps = AL(NC); float* vtmp = AL((size_t)3*NC);
  float* gsA = AL(NC); float* gsB = AL(NC);
  float* gvA = AL((size_t)3*NC); float* gvB = AL((size_t)3*NC);
  float* gtmp = AL(NC); float* gvtmp = AL((size_t)3*NC);
  float* gfs = AL(NC); float* gfv = AL((size_t)3*NC);
  float* gags = AL(NC); float* gagv = AL((size_t)3*NC);
  float* gsu = AL(NC); float* gvu = AL((size_t)3*NC);

  auto gemm = [&](const float* A, const float* B, float* C, int M, int Nt, int Kt, int flags){
    dim3 g((M + 127)/128);
    if      (Nt ==  64 && Kt ==  32) k_wmma_gemm< 64, 32><<<g, 256, 0, stream>>>(A, B, C, M, flags);
    else if (Nt ==  64 && Kt ==  64) k_wmma_gemm< 64, 64><<<g, 256, 0, stream>>>(A, B, C, M, flags);
    else if (Nt == 256 && Kt ==  64) k_wmma_gemm<256, 64><<<g, 256, 0, stream>>>(A, B, C, M, flags);
    else if (Nt ==  64 && Kt == 256) k_wmma_gemm< 64,256><<<g, 256, 0, stream>>>(A, B, C, M, flags);
    else if (Nt ==  32 && Kt ==  64) k_wmma_gemm< 32, 64><<<g, 256, 0, stream>>>(A, B, C, M, flags);
  };
  auto g1 = [&](int n){ return dim3((n + T - 1)/T); };

  // ---------------------------- Forward ----------------------------
  k_pad_w1<<<g1(32*64), T, 0, stream>>>(Wr1[0], Wr1p[0]);
  k_pad_w1<<<g1(32*64), T, 0, stream>>>(Wr1[1], Wr1p[1]);
  k_edge_geom<<<g1(E), T, 0, stream>>>(pos, shifts, ei, y1, len, ef, E);
  k_embed<<<g1(NC), T, 0, stream>>>(sp, Wemb, s0, N);
  k_e0<<<g1(N), T, 0, stream>>>(sp, batch, AE, contrib, N);
  hipMemsetAsync(v0, 0, (size_t)3*NC*sizeof(float), stream);

  float* sin_ = s0; float* vin = v0;
  for (int li = 0; li < 2; ++li) {
    int last = (li == 1);
    float* sout = last ? s2 : s1;
    float* vout = v1;
    gemm(sin_, Wup0[li], su[li], N, 64, 64, 0);
    for (int x = 0; x < 3; ++x)
      gemm(vin + (size_t)x*NC, Wup1[li], vu[li] + (size_t)x*NC, N, 64, 64, 0);
    // radial MLP
    gemm(ef, Wr1p[li], z1, E, 64, 32, 0);
    k_silu_fwd<<<g1(E*64), T, 0, stream>>>(z1, a1, E*64);
    gemm(a1, Wr2[li], z2, E, 64, 64, 0);
    k_silu_fwd<<<g1(E*64), T, 0, stream>>>(z2, a2, E*64);
    gemm(a2, Wr3[li], wb[li], E, 256, 64, 0);
    // messages + scatter
    hipMemsetAsync(ags, 0, (size_t)NC*sizeof(float), stream);
    hipMemsetAsync(agv, 0, (size_t)3*NC*sizeof(float), stream);
    k_message_fwd<<<g1(E*64), T, 0, stream>>>(ei, su[li], vu[li], y1, wb[li], ags, agv, E, NC);
    gemm(ags, Wo0[li], fsb[li], N, 64, 64, 0);
    for (int x = 0; x < 3; ++x)
      gemm(agv + (size_t)x*NC, Wo1[li], fvb[li] + (size_t)x*NC, N, 64, 64, 0);
    // product basis + skips
    k_product_fwd<<<g1(NC), T, 0, stream>>>(sp, fsb[li], fvb[li], Wps[li], Wpv[li],
                                            tmps, vtmp, N, NC, last);
    gemm(tmps, Wpl0[li], sout, N, 64, 64, 0);
    k_sc_add_s<<<g1(NC), T, 0, stream>>>(sp, sin_, Wsc0[li], sout, N);
    if (!last) {
      for (int x = 0; x < 3; ++x)
        gemm(vtmp + (size_t)x*NC, Wpl1[li], vout + (size_t)x*NC, N, 64, 64, 0);
      k_sc_add_v<<<g1(NC), T, 0, stream>>>(sp, vin, Wsc1[li], vout, N, NC);
      k_readout_lin<<<g1(N), T, 0, stream>>>(sout, WroL, batch, contrib, 1, N);
    } else {
      k_readout_nl<<<g1(N), T, 0, stream>>>(sout, WroA, WroB, batch, contrib, 2, N);
    }
    sin_ = sout; vin = vout;
  }
  k_total<<<1, 32, 0, stream>>>(total, contrib, G);

  // ---------------------------- Backward ----------------------------
  hipMemsetAsync(defb, 0, (size_t)E*32*sizeof(float), stream);
  hipMemsetAsync(dy1b, 0, (size_t)E*3*sizeof(float), stream);
  hipMemsetAsync(gsA, 0, (size_t)NC*sizeof(float), stream);
  hipMemsetAsync(gvA, 0, (size_t)3*NC*sizeof(float), stream);
  float* gs = gsA; float* gv = gvA; float* gsN = gsB; float* gvN = gvB;

  for (int li = 1; li >= 0; --li) {
    int last = (li == 1);
    float* sout = li ? s2 : s1;
    if (last) k_readout_nl_bwd<<<g1(N), T, 0, stream>>>(sout, WroA, WroB, gs, N);
    else      k_readout_lin_bwd<<<g1(NC), T, 0, stream>>>(WroL, gs, N);
    hipMemsetAsync(gsN, 0, (size_t)NC*sizeof(float), stream);
    hipMemsetAsync(gvN, 0, (size_t)3*NC*sizeof(float), stream);
    // s_new = tmp_s@Wpl0 + sc_s
    gemm(gs, Wpl0[li], gtmp, N, 64, 64, GF_TRANSB);
    k_sc_bwd_s<<<g1(NC), T, 0, stream>>>(sp, gs, Wsc0[li], gsN, N);
    if (!last) {
      for (int x = 0; x < 3; ++x)
        gemm(gv + (size_t)x*NC, Wpl1[li], gvtmp + (size_t)x*NC, N, 64, 64, GF_TRANSB);
      k_sc_bwd_v<<<g1(NC), T, 0, stream>>>(sp, gv, Wsc1[li], gvN, N, NC);
    }
    k_product_bwd<<<g1(NC), T, 0, stream>>>(sp, gtmp, gvtmp, fsb[li], fvb[li],
                                            Wps[li], Wpv[li], gfs, gfv, N, NC, last);
    gemm(gfs, Wo0[li], gags, N, 64, 64, GF_TRANSB);
    for (int x = 0; x < 3; ++x)
      gemm(gfv + (size_t)x*NC, Wo1[li], gagv + (size_t)x*NC, N, 64, 64, GF_TRANSB);
    // scatter/message backward
    hipMemsetAsync(gsu, 0, (size_t)NC*sizeof(float), stream);
    hipMemsetAsync(gvu, 0, (size_t)3*NC*sizeof(float), stream);
    k_message_bwd<<<g1(E*64), T, 0, stream>>>(ei, gags, gagv, su[li], vu[li], y1, wb[li],
                                              gsu, gvu, dwb, dy1b, E, NC);
    // radial MLP backward (recompute z1, z2; then transposed WMMA GEMMs)
    gemm(ef, Wr1p[li], z1, E, 64, 32, 0);
    k_silu_fwd<<<g1(E*64), T, 0, stream>>>(z1, a1, E*64);
    gemm(a1, Wr2[li], z2, E, 64, 64, 0);
    gemm(dwb, Wr3[li], a2, E, 64, 256, GF_TRANSB);                 // a2 := da2
    k_silu_bwd<<<g1(E*64), T, 0, stream>>>(z2, a2, a2, E*64);      // a2 := dz2
    gemm(a2, Wr2[li], a1, E, 64, 64, GF_TRANSB);                   // a1 := da1
    k_silu_bwd<<<g1(E*64), T, 0, stream>>>(z1, a1, a1, E*64);      // a1 := dz1
    gemm(a1, Wr1p[li], defb, E, 32, 64, GF_TRANSB | GF_ACCUM);     // def += dz1@Wr1p^T
    // sender grads -> previous layer s,v
    gemm(gsu, Wup0[li], gsN, N, 64, 64, GF_TRANSB | GF_ACCUM);
    for (int x = 0; x < 3; ++x)
      gemm(gvu + (size_t)x*NC, Wup1[li], gvN + (size_t)x*NC, N, 64, 64, GF_TRANSB | GF_ACCUM);
    // rotate grad buffers
    float* ts = gs; gs = gsN; gsN = ts;
    float* tv = gv; gv = gvN; gvN = tv;
  }

  // edge geometry backward -> forces
  k_edge_geom_bwd<<<g1(E), T, 0, stream>>>(ei, y1, len, defb, dy1b, forces, E);
}
